// GATv2_41635412967544
// MI455X (gfx1250) — matrix operand order repaired and checked by reading, not compiled, not run on hardware
//
#include <hip/hip_runtime.h>
#include <hip/hip_bf16.h>

// ---------------------------------------------------------------------------
// GATv2 (3 layers) for MI455X / gfx1250.
//  - Dense projections: bf16 WMMA (v_wmma_f32_16x16x32_bf16), f32 accumulate.
//    Each wave computes 16 rows x (CT*16) cols: A fragment loaded once per
//    k-step and reused across CT WMMAs (4x less A traffic than 1 tile/wave).
//  - Edge softmax/aggregation: 3 atomic passes (max, exp-sum, weighted scatter).
// ---------------------------------------------------------------------------

typedef __attribute__((ext_vector_type(16))) __bf16 v16bf;
typedef __attribute__((ext_vector_type(8)))  __bf16 v8bf;
typedef __attribute__((ext_vector_type(8)))  float  v8f;
typedef __attribute__((ext_vector_type(4)))  float  v4f;

#define SLOPE 0.2f

// ---------------------------------------------------------------------------
// GEMM: C[M x Ncols] = A[M x K](f32, row-major) * Wt[Ncols x K](bf16)^T
// blockDim = (32, 8): 8 waves, each wave owns one 16-row tile and CT 16-col
// tiles. K multiple of 32. Fragment layouts per ISA 7.12.2.
// ---------------------------------------------------------------------------
template <int CT>
__global__ __launch_bounds__(256)
void gat_gemm_wmma_bf16(const float* __restrict__ A,
                        const __bf16* __restrict__ Bt,
                        float* __restrict__ C,
                        int M, int K, int ldc) {
  const int lane = threadIdx.x;        // 0..31 (wave32)
  const int wave = threadIdx.y;        // 0..7
  const int half = lane >> 4;          // 0: lanes 0-15, 1: lanes 16-31
  const int l16  = lane & 15;

  const int rowBase = (blockIdx.x * 8 + wave) * 16;
  if (rowBase >= M) return;            // wave-uniform: EXEC stays all-ones
  const int colBase = blockIdx.y * (CT * 16);

  const float* arow = A + (size_t)(rowBase + l16) * K;   // A row for this lane
  const __bf16* brow[CT];
#pragma unroll
  for (int t = 0; t < CT; ++t)         // output column (row of Wt) per tile
    brow[t] = Bt + (size_t)(colBase + t * 16 + l16) * K;

  v8f acc[CT];
#pragma unroll
  for (int t = 0; t < CT; ++t) acc[t] = (v8f){};

  for (int k0 = 0; k0 < K; k0 += 32) {
    // A fragment (16x32 bf16): lanes<16 hold K = k0+[0..7] and k0+[16..23];
    // lanes>=16 hold K = k0+[8..15] and k0+[24..31]. Convert f32->bf16 here.
    const int ka0 = k0 + half * 8;
    const int ka1 = ka0 + 16;
    v4f a00 = *(const v4f*)(arow + ka0);
    v4f a01 = *(const v4f*)(arow + ka0 + 4);
    v4f a10 = *(const v4f*)(arow + ka1);
    v4f a11 = *(const v4f*)(arow + ka1 + 4);
    v16bf av;
#pragma unroll
    for (int i = 0; i < 4; ++i) {
      av[i]      = (__bf16)a00[i];
      av[4 + i]  = (__bf16)a01[i];
      av[8 + i]  = (__bf16)a10[i];
      av[12 + i] = (__bf16)a11[i];
    }

    if (k0 + 32 < K) __builtin_prefetch(arow + k0 + 32, 0, 1);  // global_prefetch_b8

    // B fragments (32x16 bf16): lane's column = colBase+t*16+l16;
    // lanes<16 hold K = k0+[0..7] and k0+[16..23]; lanes>=16 hold K = k0+[8..15] and k0+[24..31].
    const int kb = k0 + half * 8;
#pragma unroll
    for (int t = 0; t < CT; ++t) {
      v8bf b0 = *(const v8bf*)(brow[t] + kb);
      v8bf b1 = *(const v8bf*)(brow[t] + kb + 16);
      v16bf bv;
#pragma unroll
      for (int i = 0; i < 8; ++i) { bv[i] = b0[i]; bv[8 + i] = b1[i]; }
      // D = A*B + C  (8 args: neg_a, A, neg_b, B, c_mod, C, reuse_a, reuse_b)
      acc[t] = __builtin_amdgcn_wmma_f32_16x16x32_bf16(false, av, false, bv,
                                                       (short)0, acc[t],
                                                       false, false);
    }
  }

  // C/D layout: VGPR r -> M = r + 8*half, N = l16.
#pragma unroll
  for (int t = 0; t < CT; ++t) {
    float* crow = C + (size_t)(rowBase + 8 * half) * ldc + colBase + t * 16 + l16;
#pragma unroll
    for (int r = 0; r < 8; ++r) crow[(size_t)r * ldc] = acc[t][r];
  }
}

// ---------------------------------------------------------------------------
// W[K x Nout] (f32, row-major) -> Wt[Nout x K] (bf16)
// ---------------------------------------------------------------------------
__global__ void gat_transpose_bf16(const float* __restrict__ W,
                                   __bf16* __restrict__ Wt,
                                   int K, int Nout) {
  int i = blockIdx.x * blockDim.x + threadIdx.x;
  if (i >= K * Nout) return;
  int n = i / K, k = i - n * K;
  Wt[i] = (__bf16)W[(size_t)k * Nout + n];
}

__global__ void gat_fill_f32(float* __restrict__ p, long n, float v) {
  long i = (long)blockIdx.x * blockDim.x + threadIdx.x;
  if (i < n) p[i] = v;
}

__device__ inline float lrelu(float x) { return x > 0.f ? x : SLOPE * x; }

__device__ inline void atomicMaxF(float* addr, float val) {
  unsigned int* u = (unsigned int*)addr;
  unsigned int cur = *u;
  while (__uint_as_float(cur) < val) {
    unsigned int prev = atomicCAS(u, cur, __float_as_uint(val));
    if (prev == cur) break;
    cur = prev;
  }
}

// ---------------------------------------------------------------------------
// Pass 1: per (edge, head) score = dot(leaky(f[src]+f[dst]), a[h]); atomic max per (dst,h)
// ---------------------------------------------------------------------------
__global__ void gat_edge_scores(const float* __restrict__ f,
                                const int* __restrict__ src,
                                const int* __restrict__ dst,
                                const float* __restrict__ a,
                                float* __restrict__ esc,
                                float* __restrict__ mmax,
                                long E, int heads, int dim) {
  long t = (long)blockIdx.x * blockDim.x + threadIdx.x;
  if (t >= E * heads) return;
  int e = (int)(t / heads), h = (int)(t - (long)e * heads);
  const float* fs = f + (size_t)src[e] * heads * dim + (size_t)h * dim;
  const float* fd = f + (size_t)dst[e] * heads * dim + (size_t)h * dim;
  const float* ah = a + (size_t)h * dim;
  float s = 0.f;
  for (int d = 0; d < dim; d += 4) {
    float4 v1 = *(const float4*)(fs + d);
    float4 v2 = *(const float4*)(fd + d);
    float4 av = *(const float4*)(ah + d);
    s += lrelu(v1.x + v2.x) * av.x;
    s += lrelu(v1.y + v2.y) * av.y;
    s += lrelu(v1.z + v2.z) * av.z;
    s += lrelu(v1.w + v2.w) * av.w;
  }
  esc[t] = s;
  atomicMaxF(&mmax[(size_t)dst[e] * heads + h], s);
}

// ---------------------------------------------------------------------------
// Pass 2: ex = exp(s - max); atomic sum per (dst,h)
// ---------------------------------------------------------------------------
__global__ void gat_edge_expsum(const int* __restrict__ dst,
                                const float* __restrict__ mmax,
                                float* __restrict__ esc,
                                float* __restrict__ z,
                                long E, int heads) {
  long t = (long)blockIdx.x * blockDim.x + threadIdx.x;
  if (t >= E * heads) return;
  int e = (int)(t / heads), h = (int)(t - (long)e * heads);
  size_t di = (size_t)dst[e] * heads + h;
  float ex = __expf(esc[t] - mmax[di]);
  esc[t] = ex;
  atomicAdd(&z[di], ex);
}

// ---------------------------------------------------------------------------
// Pass 3: rst[dst,h,:] += (ex/z) * f[src,h,:]   (4 dims per thread)
// ---------------------------------------------------------------------------
__global__ void gat_edge_aggregate(const float* __restrict__ esc,
                                   const float* __restrict__ z,
                                   const float* __restrict__ f,
                                   const int* __restrict__ src,
                                   const int* __restrict__ dst,
                                   float* __restrict__ rst,
                                   long E, int heads, int dim) {
  const int chunks = dim >> 2;
  long t = (long)blockIdx.x * blockDim.x + threadIdx.x;
  if (t >= E * heads * chunks) return;
  int c  = (int)(t % chunks);
  long eh = t / chunks;
  int h = (int)(eh % heads);
  int e = (int)(eh / heads);
  int d = dst[e];
  float alpha = esc[eh] / z[(size_t)d * heads + h];
  const float* fsrc = f + (size_t)src[e] * heads * dim + (size_t)h * dim + (size_t)c * 4;
  float*       out  = rst + (size_t)d * heads * dim + (size_t)h * dim + (size_t)c * 4;
  float4 v = *(const float4*)fsrc;
  atomicAdd(out + 0, alpha * v.x);
  atomicAdd(out + 1, alpha * v.y);
  atomicAdd(out + 2, alpha * v.z);
  atomicAdd(out + 3, alpha * v.w);
}

// ---------------------------------------------------------------------------
// Epilogue: out = elu(rst [+ res])
// ---------------------------------------------------------------------------
__global__ void gat_epilogue_elu(const float* __restrict__ rst,
                                 const float* __restrict__ res,
                                 float* __restrict__ out, long n) {
  long i = (long)blockIdx.x * blockDim.x + threadIdx.x;
  if (i >= n) return;
  float v = rst[i] + (res ? res[i] : 0.f);
  out[i] = v > 0.f ? v : __expf(v) - 1.f;
}

__global__ void gat_add_inplace(float* __restrict__ out,
                                const float* __restrict__ res, long n) {
  long i = (long)blockIdx.x * blockDim.x + threadIdx.x;
  if (i < n) out[i] += res[i];
}

// ---------------------------------------------------------------------------
// Host-side orchestration
// ---------------------------------------------------------------------------
static inline long cdivl(long a, long b) { return (a + b - 1) / b; }

extern "C" void kernel_launch(void* const* d_in, const int* in_sizes, int n_in,
                              void* d_out, int out_size, void* d_ws, size_t ws_size,
                              hipStream_t stream) {
  (void)n_in; (void)out_size; (void)ws_size;
  const float* x     = (const float*)d_in[0];
  const int*   src   = (const int*)  d_in[1];
  const int*   dst   = (const int*)  d_in[2];
  const float* W0    = (const float*)d_in[3];
  const float* a0    = (const float*)d_in[4];
  const float* W1    = (const float*)d_in[5];
  const float* a1    = (const float*)d_in[6];
  const float* W2    = (const float*)d_in[7];
  const float* a2    = (const float*)d_in[8];
  const float* Wres2 = (const float*)d_in[9];

  const int IN = 256, HD = 256, H = 4, D = 64, C = 32;
  const long N = in_sizes[0] / IN;      // 50000
  const long E = in_sizes[1];           // 850000

  // Workspace carve-up (256B aligned slices)
  char* ws = (char*)d_ws;
  size_t off = 0;
  auto carve = [&](size_t bytes) -> void* {
    void* p = ws + off;
    off += (bytes + 255) & ~(size_t)255;
    return p;
  };
  float*  P    = (float*) carve((size_t)N * HD * 4);   // projected features f
  float*  Q    = (float*) carve((size_t)N * HD * 4);   // aggregation accumulator
  float*  R    = (float*) carve((size_t)N * HD * 4);   // layer input h (f32)
  float*  esc  = (float*) carve((size_t)E * H  * 4);   // edge scores / exp
  float*  mmax = (float*) carve((size_t)N * H  * 4);
  float*  zsum = (float*) carve((size_t)N * H  * 4);
  __bf16* Wt0  = (__bf16*)carve((size_t)HD * IN * 2);  // [256 out][256 in]
  __bf16* Wt1  = (__bf16*)carve((size_t)HD * HD * 2);
  __bf16* Wt2  = (__bf16*)carve((size_t)C  * HD * 2);  // [32][256]
  __bf16* Wrt  = (__bf16*)carve((size_t)C  * HD * 2);
  float*  F2   = (float*) carve((size_t)N * C * 4);
  float*  RES2 = (float*) carve((size_t)N * C * 4);
  float*  OUT  = (float*)d_out;                        // [N, 32]

  const int TB = 256;
  const dim3 gblk(32, 8);                              // 8 waves per block
  const unsigned gRows = (unsigned)cdivl(N, 128);      // 8 row tiles per block

  // --- Weight transposes to bf16 (tiny; stay L2-resident for all GEMM tiles)
  gat_transpose_bf16<<<cdivl(IN * HD, TB), TB, 0, stream>>>(W0, Wt0, IN, HD);
  gat_transpose_bf16<<<cdivl(HD * HD, TB), TB, 0, stream>>>(W1, Wt1, HD, HD);
  gat_transpose_bf16<<<cdivl(HD * C,  TB), TB, 0, stream>>>(W2, Wt2, HD, C);
  gat_transpose_bf16<<<cdivl(HD * C,  TB), TB, 0, stream>>>(Wres2, Wrt, HD, C);

  // =========================== Layer 0 ===========================
  {
    dim3 g(gRows, HD / 64);   // CT=4 -> 64 cols per wave
    gat_gemm_wmma_bf16<4><<<g, gblk, 0, stream>>>(x, Wt0, P, (int)N, IN, HD);

    gat_fill_f32<<<cdivl(N * H,  TB), TB, 0, stream>>>(mmax, N * H, -3.0e38f);
    gat_fill_f32<<<cdivl(N * H,  TB), TB, 0, stream>>>(zsum, N * H, 0.f);
    gat_fill_f32<<<cdivl(N * HD, TB), TB, 0, stream>>>(Q, N * HD, 0.f);

    gat_edge_scores   <<<cdivl(E * H, TB), TB, 0, stream>>>(P, src, dst, a0, esc, mmax, E, H, D);
    gat_edge_expsum   <<<cdivl(E * H, TB), TB, 0, stream>>>(dst, mmax, esc, zsum, E, H);
    gat_edge_aggregate<<<cdivl(E * H * (D / 4), TB), TB, 0, stream>>>(esc, zsum, P, src, dst, Q, E, H, D);

    gat_epilogue_elu<<<cdivl(N * HD, TB), TB, 0, stream>>>(Q, nullptr, R, N * HD);
  }

  // =========================== Layer 1 ===========================
  {
    dim3 g(gRows, HD / 64);
    gat_gemm_wmma_bf16<4><<<g, gblk, 0, stream>>>(R, Wt1, P, (int)N, HD, HD);

    gat_fill_f32<<<cdivl(N * H,  TB), TB, 0, stream>>>(mmax, N * H, -3.0e38f);
    gat_fill_f32<<<cdivl(N * H,  TB), TB, 0, stream>>>(zsum, N * H, 0.f);
    gat_fill_f32<<<cdivl(N * HD, TB), TB, 0, stream>>>(Q, N * HD, 0.f);

    gat_edge_scores   <<<cdivl(E * H, TB), TB, 0, stream>>>(P, src, dst, a1, esc, mmax, E, H, D);
    gat_edge_expsum   <<<cdivl(E * H, TB), TB, 0, stream>>>(dst, mmax, esc, zsum, E, H);
    gat_edge_aggregate<<<cdivl(E * H * (D / 4), TB), TB, 0, stream>>>(esc, zsum, P, src, dst, Q, E, H, D);

    // residual (identity reshape) + elu, written back over R
    gat_epilogue_elu<<<cdivl(N * HD, TB), TB, 0, stream>>>(Q, R, R, N * HD);
  }

  // =========================== Layer 2 ===========================
  {
    dim3 g(gRows, 1);         // CT=2 -> all 32 cols in one wave pass
    gat_gemm_wmma_bf16<2><<<g, gblk, 0, stream>>>(R, Wt2, F2,   (int)N, HD, C);
    gat_gemm_wmma_bf16<2><<<g, gblk, 0, stream>>>(R, Wrt, RES2, (int)N, HD, C);

    gat_fill_f32<<<cdivl(N,     TB), TB, 0, stream>>>(mmax, N, -3.0e38f);
    gat_fill_f32<<<cdivl(N,     TB), TB, 0, stream>>>(zsum, N, 0.f);
    gat_fill_f32<<<cdivl(N * C, TB), TB, 0, stream>>>(OUT, N * C, 0.f);

    gat_edge_scores   <<<cdivl(E, TB), TB, 0, stream>>>(F2, src, dst, a2, esc, mmax, E, 1, C);
    gat_edge_expsum   <<<cdivl(E, TB), TB, 0, stream>>>(dst, mmax, esc, zsum, E, 1);
    gat_edge_aggregate<<<cdivl(E * (C / 4), TB), TB, 0, stream>>>(esc, zsum, F2, src, dst, OUT, E, 1, C);

    // + res_fc residual; no activation; mean over 1 head == identity
    gat_add_inplace<<<cdivl(N * C, TB), TB, 0, stream>>>(OUT, RES2, N * C);
  }
}
